// BasePointSAModule_60344290509346
// MI455X (gfx1250) — matrix operand order repaired and checked by reading, not compiled, not run on hardware
//
#include <hip/hip_runtime.h>
#include <math.h>

// ---------------- problem constants ----------------
#define B_   4
#define N_   16384
#define C_   64
#define M_   1024
#define K_   32
#define CP_  67          // C + 3
#define CPP_ 96          // CP padded to multiple of 32 for bf16 WMMA K-depth
#define O0_  128
#define O1_  256
#define RAD2 0.25f       // RADIUS^2

typedef __bf16 bf16_t;
typedef __attribute__((ext_vector_type(16))) __bf16 v16bf;
typedef __attribute__((ext_vector_type(8)))  __bf16 v8bf;
typedef __attribute__((ext_vector_type(8)))  float  v8f;

// ---------------- workspace layout (bytes) ----------------
#define OFF_FEATT 0u                        // B*N*C f32          = 16,777,216
#define OFF_GWS   16777216u                 // B*M*K*96 bf16      = 25,165,824
#define OFF_PE    41943040u                 // M*CP f32           =    274,432
#define OFF_PW    42217472u                 // B*M f32            =     16,384
#define OFF_EMB   42233856u                 // B*CP f32 (pad 4096)
#define OFF_W0P   42237952u                 // 24 frags * 512 bf16 =    24,576
#define OFF_W1P   42262528u                 // 64 frags * 512 bf16 =    65,536

// WMMA A-fragment K mapping (contiguous: k = 16*(lane/16) + j, matches B staging)
static __device__ __forceinline__ int kmapA(int lane, int j) {
  return ((lane < 16) ? 0 : 16) + j;
}
// fragment-major load: lane's 16 bf16 contiguous -> 2x b128 loads
static __device__ __forceinline__ v16bf load_frag(const bf16_t* buf, int f, int lane) {
  return *(const v16bf*)(buf + ((f * 32 + lane) << 4));
}
static __device__ __forceinline__ v8f wmma_bf16(v16bf a, v16bf b, v8f c) {
  return __builtin_amdgcn_wmma_f32_16x16x32_bf16(false, a, false, b, (short)0, c, false, false);
}

// ---------------- kernel 0: weight pack (fragment-major bf16) + emb zero --
__global__ void prep_kernel(const float* __restrict__ w0, const float* __restrict__ w1,
                            bf16_t* __restrict__ w0p, bf16_t* __restrict__ w1p,
                            float* __restrict__ emb_sum) {
  int t = blockIdx.x * blockDim.x + threadIdx.x;
  if (t < O0_ * CPP_) {                       // 24 fragments (8 ot x 3 ks) * 512
    int f = t >> 9, rem = t & 511, lane = rem >> 4, j = rem & 15;
    int ot = f / 3, ks = f % 3;
    int row = ot * 16 + (lane & 15);
    int k = ks * 32 + kmapA(lane, j);
    w0p[t] = (bf16_t)((k < CP_) ? w0[row * CP_ + k] : 0.0f);
  }
  int t2 = t - O0_ * CPP_;
  if (t2 >= 0 && t2 < O1_ * O0_) {            // 64 fragments (16 ot2 x 4 ks) * 512
    int f = t2 >> 9, rem = t2 & 511, lane = rem >> 4, j = rem & 15;
    int ot2 = f / 4, ks = f % 4;
    int row = ot2 * 16 + (lane & 15);
    int k = ks * 32 + kmapA(lane, j);
    w1p[t2] = (bf16_t)w1[row * O0_ + k];
  }
  int t3 = t - O0_ * CPP_ - O1_ * O0_;
  if (t3 >= 0 && t3 < B_ * CP_) emb_sum[t3] = 0.0f;
}

// ---------------- kernel 1: features (B,C,N) -> featT (B,N,C) ------------
__global__ __launch_bounds__(1024) void transpose_kernel(const float* __restrict__ f,
                                                         float* __restrict__ ft) {
  __shared__ float tile[32][33];
  int b = blockIdx.z;
  int n0 = blockIdx.x * 32, c0 = blockIdx.y * 32;
  int tx = threadIdx.x, ty = threadIdx.y;
  tile[ty][tx] = f[((size_t)b * C_ + (c0 + ty)) * N_ + (n0 + tx)];
  __syncthreads();
  ft[((size_t)b * N_ + (n0 + ty)) * C_ + (c0 + tx)] = tile[tx][ty];
}

// ---------------- kernel 2: furthest point sampling ----------------------
__global__ __launch_bounds__(1024) void fps_kernel(const float* __restrict__ xyz,
                                                   float* __restrict__ out_xyz,
                                                   float* __restrict__ out_idx) {
  int b = blockIdx.x, t = threadIdx.x;
  int lane = t & 31, wid = t >> 5;
  __shared__ float swv[32];
  __shared__ int   swi[32];
  __shared__ float scx, scy, scz;
  const float* base = xyz + (size_t)b * N_ * 3;

  float px[16], py[16], pz[16], md[16];
#pragma unroll
  for (int j = 0; j < 16; ++j) {
    int i = j * 1024 + t;
    px[j] = base[i * 3 + 0]; py[j] = base[i * 3 + 1]; pz[j] = base[i * 3 + 2];
    md[j] = 1e10f;
  }
  float cx = base[0], cy = base[1], cz = base[2];
  if (t == 0) {
    out_idx[b * M_] = 0.0f;
    out_xyz[(size_t)(b * M_) * 3 + 0] = cx;
    out_xyz[(size_t)(b * M_) * 3 + 1] = cy;
    out_xyz[(size_t)(b * M_) * 3 + 2] = cz;
  }
  for (int s = 1; s < M_; ++s) {
    float bv = -1.0f; int bi = 0x7fffffff;
#pragma unroll
    for (int j = 0; j < 16; ++j) {
      float dx = px[j] - cx, dy = py[j] - cy, dz = pz[j] - cz;
      float d = dx * dx + dy * dy + dz * dz;
      md[j] = fminf(md[j], d);
      int i = j * 1024 + t;
      if (md[j] > bv || (md[j] == bv && i < bi)) { bv = md[j]; bi = i; }
    }
#pragma unroll
    for (int off = 16; off > 0; off >>= 1) {
      float ov = __shfl_xor(bv, off, 32);
      int   oi = __shfl_xor(bi, off, 32);
      if (ov > bv || (ov == bv && oi < bi)) { bv = ov; bi = oi; }
    }
    if (lane == 0) { swv[wid] = bv; swi[wid] = bi; }
    __syncthreads();
    if (wid == 0) {
      bv = swv[lane]; bi = swi[lane];
#pragma unroll
      for (int off = 16; off > 0; off >>= 1) {
        float ov = __shfl_xor(bv, off, 32);
        int   oi = __shfl_xor(bi, off, 32);
        if (ov > bv || (ov == bv && oi < bi)) { bv = ov; bi = oi; }
      }
      if (lane == 0) {
        scx = base[bi * 3 + 0]; scy = base[bi * 3 + 1]; scz = base[bi * 3 + 2];
        out_idx[b * M_ + s] = (float)bi;
        out_xyz[(size_t)(b * M_ + s) * 3 + 0] = scx;
        out_xyz[(size_t)(b * M_ + s) * 3 + 1] = scy;
        out_xyz[(size_t)(b * M_ + s) * 3 + 2] = scz;
      }
    }
    __syncthreads();
    cx = scx; cy = scy; cz = scz;
  }
}

// ---------------- kernel 3: ball query + gather + emb accumulate ---------
// gws layout: [b][m][k][CPP_] bf16 (zero padded 67..95), row = 192B per (m,k)
__global__ __launch_bounds__(256) void ballq_kernel(const float* __restrict__ xyz,
                                                    const float* __restrict__ featT,
                                                    const float* __restrict__ new_xyz,
                                                    bf16_t* __restrict__ gws,
                                                    float* __restrict__ emb_sum) {
  int b = blockIdx.y;
  int w = threadIdx.x >> 5;
  int lane = threadIdx.x & 31;
  int m = blockIdx.x * 8 + w;
  __shared__ int sidx[8][K_];
  const float* base = xyz + (size_t)b * N_ * 3;
  float cx = new_xyz[(size_t)(b * M_ + m) * 3 + 0];
  float cy = new_xyz[(size_t)(b * M_ + m) * 3 + 1];
  float cz = new_xyz[(size_t)(b * M_ + m) * 3 + 2];

  // first K hits in ascending index order (== top_k(-scores) semantics)
  int filled = 0;
  for (int b0 = 0; b0 < N_ && filled < K_; b0 += 32) {
    int i = b0 + lane;
    float dx = base[i * 3 + 0] - cx;
    float dy = base[i * 3 + 1] - cy;
    float dz = base[i * 3 + 2] - cz;
    bool hit = (dx * dx + dy * dy + dz * dz) <= RAD2;
    unsigned msk = __builtin_amdgcn_ballot_w32(hit);
    int pre = __popc(msk & ((1u << lane) - 1u));
    if (hit && (filled + pre) < K_) sidx[w][filled + pre] = i;
    filled += __popc(msk);
  }
  int total = filled < K_ ? filled : K_;
  __syncthreads();
  int iSel = (lane < total) ? sidx[w][lane] : sidx[w][0];  // pad with first hit

  // gather: lane == k ; features contiguous 256B per point
  float gx = base[iSel * 3 + 0] - cx;
  float gy = base[iSel * 3 + 1] - cy;
  float gz = base[iSel * 3 + 2] - cz;
  const float4* fb4 = (const float4*)(featT + ((size_t)b * N_ + iSel) * C_);
  float4 f4[16];
#pragma unroll
  for (int i = 0; i < 16; ++i) f4[i] = fb4[i];

  float fa[C_];
#pragma unroll
  for (int i = 0; i < 16; ++i) {
    fa[4 * i + 0] = f4[i].x; fa[4 * i + 1] = f4[i].y;
    fa[4 * i + 2] = f4[i].z; fa[4 * i + 3] = f4[i].w;
  }

  bf16_t row[CPP_];
  row[0] = (bf16_t)gx; row[1] = (bf16_t)gy; row[2] = (bf16_t)gz;
#pragma unroll
  for (int c = 0; c < C_; ++c) row[3 + c] = (bf16_t)fa[c];
#pragma unroll
  for (int c = CP_; c < CPP_; ++c) row[c] = (bf16_t)0.0f;

  bf16_t* grow = gws + ((size_t)((b * M_ + m) * K_) + lane) * CPP_;
#pragma unroll
  for (int i = 0; i < 12; ++i) {              // 12 x 16B packed stores
    v8bf v;
#pragma unroll
    for (int j = 0; j < 8; ++j) v[j] = row[8 * i + j];
    *(v8bf*)(grow + 8 * i) = v;
  }

  // emb accumulation (f32 wave reductions + one atomic per channel)
  float s;
  s = gx; for (int o = 16; o > 0; o >>= 1) s += __shfl_xor(s, o, 32);
  if (lane == 0) atomicAdd(&emb_sum[b * CP_ + 0], s);
  s = gy; for (int o = 16; o > 0; o >>= 1) s += __shfl_xor(s, o, 32);
  if (lane == 0) atomicAdd(&emb_sum[b * CP_ + 1], s);
  s = gz; for (int o = 16; o > 0; o >>= 1) s += __shfl_xor(s, o, 32);
  if (lane == 0) atomicAdd(&emb_sum[b * CP_ + 2], s);
#pragma unroll
  for (int c = 0; c < C_; ++c) {
    s = fa[c];
    for (int o = 16; o > 0; o >>= 1) s += __shfl_xor(s, o, 32);
    if (lane == 0) atomicAdd(&emb_sum[b * CP_ + 3 + c], s);
  }
}

// ---------------- kernel 4: prompt proj + LayerNorm + exact GELU ---------
__global__ __launch_bounds__(256) void pe_kernel(const float* __restrict__ prompt,
                                                 const float* __restrict__ proj_w,
                                                 const float* __restrict__ proj_b,
                                                 const float* __restrict__ ln_g,
                                                 const float* __restrict__ ln_b,
                                                 float* __restrict__ pe) {
  int m = blockIdx.x * 8 + (threadIdx.x >> 5);
  int lane = threadIdx.x & 31;
  float p0 = prompt[m * 2 + 0], p1 = prompt[m * 2 + 1];
  float v[3]; int cs[3] = {lane, lane + 32, lane + 64};
  float s = 0.0f;
#pragma unroll
  for (int j = 0; j < 3; ++j) {
    v[j] = 0.0f;
    if (cs[j] < CP_) {
      v[j] = proj_w[cs[j] * 2 + 0] * p0 + proj_w[cs[j] * 2 + 1] * p1 + proj_b[cs[j]];
      s += v[j];
    }
  }
  for (int o = 16; o > 0; o >>= 1) s += __shfl_xor(s, o, 32);
  float mu = s * (1.0f / CP_);
  float q = 0.0f;
#pragma unroll
  for (int j = 0; j < 3; ++j)
    if (cs[j] < CP_) { float d = v[j] - mu; q += d * d; }
  for (int o = 16; o > 0; o >>= 1) q += __shfl_xor(q, o, 32);
  float inv = rsqrtf(q * (1.0f / CP_) + 1e-5f);
#pragma unroll
  for (int j = 0; j < 3; ++j)
    if (cs[j] < CP_) {
      float y = (v[j] - mu) * inv * ln_g[cs[j]] + ln_b[cs[j]];
      pe[m * CP_ + cs[j]] = 0.5f * y * (1.0f + erff(y * 0.70710678118654752f));
    }
}

// ---------------- kernel 5: pw = softmax(emb @ pw_w.T + pw_b) ------------
__global__ __launch_bounds__(1024) void pw_kernel(const float* __restrict__ emb_sum,
                                                  const float* __restrict__ pw_w,
                                                  const float* __restrict__ pw_b,
                                                  float* __restrict__ pw) {
  int b = blockIdx.x, t = threadIdx.x;
  int lane = t & 31, wid = t >> 5;
  __shared__ float se[CP_];
  __shared__ float red[32];
  if (t < CP_) se[t] = emb_sum[b * CP_ + t] * (1.0f / (float)(M_ * K_));
  __syncthreads();
  float acc = pw_b[t];
  for (int c = 0; c < CP_; ++c) acc += se[c] * pw_w[t * CP_ + c];

  float v = acc;
  for (int o = 16; o > 0; o >>= 1) v = fmaxf(v, __shfl_xor(v, o, 32));
  if (lane == 0) red[wid] = v;
  __syncthreads();
  if (wid == 0) {
    float x = red[lane];
    for (int o = 16; o > 0; o >>= 1) x = fmaxf(x, __shfl_xor(x, o, 32));
    if (lane == 0) red[0] = x;
  }
  __syncthreads();
  float mx = red[0];
  __syncthreads();
  float e = expf(acc - mx);
  v = e;
  for (int o = 16; o > 0; o >>= 1) v += __shfl_xor(v, o, 32);
  if (lane == 0) red[wid] = v;
  __syncthreads();
  if (wid == 0) {
    float x = red[lane];
    for (int o = 16; o > 0; o >>= 1) x += __shfl_xor(x, o, 32);
    if (lane == 0) red[0] = x;
  }
  __syncthreads();
  pw[b * M_ + t] = e / red[0];
}

// ---------------- kernel 6: fused 2-layer MLP (bf16 WMMA) + K-maxpool ----
// LDS tiles stored fragment-major: frag f -> 512 bf16, lane's 16 contiguous.
// g_l : 3 ks x 8 ct = 24 frags ; h_l : 4 ks x 8 ct = 32 frags.
__global__ __launch_bounds__(128) void mlp_kernel(const bf16_t* __restrict__ gws,
                                                  const float* __restrict__ pe,
                                                  const float* __restrict__ pw,
                                                  const bf16_t* __restrict__ w0p,
                                                  const float* __restrict__ b0,
                                                  const bf16_t* __restrict__ w1p,
                                                  const float* __restrict__ b1,
                                                  float* __restrict__ pooled) {
  int b  = blockIdx.y;
  int m0 = blockIdx.x * 4;
  int t = threadIdx.x, lane = t & 31, wid = t >> 5;

  __shared__ alignas(32) bf16_t g_l[24 * 512];   // 24 KB
  __shared__ alignas(32) bf16_t h_l[32 * 512];   // 32 KB
  __shared__ float pe_l[4 * CP_];
  __shared__ float pw_l[4];

  for (int i = t; i < 4 * CP_; i += 128)
    pe_l[i] = pe[(m0 + i / CP_) * CP_ + (i % CP_)];
  if (t < 4) pw_l[t] = pw[b * M_ + m0 + t];
  __syncthreads();

  // ---- stage g tile (pe*pw folded in), written in B-fragment layout ----
  {
    int col = t, ml = col >> 5, k = col & 31, m = m0 + ml;
    float pwv = pw_l[ml];
    const v16bf* grow = (const v16bf*)(gws + ((size_t)((b * M_ + m) * K_) + k) * CPP_);
    v16bf raw[6];
#pragma unroll
    for (int i = 0; i < 6; ++i) raw[i] = grow[i];
    int ct = col >> 4, n = col & 15;
#pragma unroll
    for (int kt = 0; kt < 3; ++kt) {
#pragma unroll
      for (int half = 0; half < 2; ++half) {
        int i = kt * 2 + half;
        v16bf out;
#pragma unroll
        for (int j = 0; j < 16; ++j) {
          int c = i * 16 + j;
          float v = 0.0f;
          if (c < CP_) v = (float)raw[i][j] + pe_l[ml * CP_ + c] * pwv;
          out[j] = (bf16_t)v;
        }
        int laneg = n + (half << 4);
        *(v16bf*)(g_l + (((kt * 8 + ct) * 32 + laneg) << 4)) = out;
      }
    }
  }
  __syncthreads();

  int jb = (lane >= 16) ? 8 : 0;

  // ---- layer 1: h = relu(W0 @ g + b0) ; B fragments held in registers ----
  {
    v16bf bg[2][3];
#pragma unroll
    for (int cti = 0; cti < 2; ++cti) {
      int ct = wid + cti * 4;
#pragma unroll
      for (int ks = 0; ks < 3; ++ks)
        bg[cti][ks] = load_frag(g_l, ks * 8 + ct, lane);
    }
    for (int ot = 0; ot < 8; ++ot) {
      v16bf a0 = load_frag(w0p, ot * 3 + 0, lane);
      v16bf a1 = load_frag(w0p, ot * 3 + 1, lane);
      v16bf a2 = load_frag(w0p, ot * 3 + 2, lane);
      float bia[8];
#pragma unroll
      for (int r = 0; r < 8; ++r) bia[r] = b0[ot * 16 + jb + r];
#pragma unroll
      for (int cti = 0; cti < 2; ++cti) {
        int ct = wid + cti * 4;
        v8f acc = {0.f, 0.f, 0.f, 0.f, 0.f, 0.f, 0.f, 0.f};
        acc = wmma_bf16(a0, bg[cti][0], acc);
        acc = wmma_bf16(a1, bg[cti][1], acc);
        acc = wmma_bf16(a2, bg[cti][2], acc);
        // rows rbase..rbase+7 of h map to 8 consecutive j slots of one h-fragment
        int lane2 = (lane & 15) + ((ot & 1) << 4);
        bf16_t* dst = h_l + (((((ot >> 1) * 8) + ct) * 32 + lane2) << 4) + jb;
        v8bf hv;
#pragma unroll
        for (int r = 0; r < 8; ++r) hv[r] = (bf16_t)fmaxf(acc[r] + bia[r], 0.0f);
        *(v8bf*)dst = hv;   // one packed 16B LDS store
      }
    }
  }
  __syncthreads();

  // ---- layer 2: relu(W1 @ h + b1), max over K=32 per m -------------------
  {
    int ml = wid;                 // one m per wave
    int mm = m0 + ml;
    v16bf bh[2][4];
#pragma unroll
    for (int cti = 0; cti < 2; ++cti)
#pragma unroll
      for (int ks = 0; ks < 4; ++ks)
        bh[cti][ks] = load_frag(h_l, ks * 8 + (2 * ml + cti), lane);

    for (int ot2 = 0; ot2 < 16; ++ot2) {
      v16bf a0 = load_frag(w1p, ot2 * 4 + 0, lane);
      v16bf a1 = load_frag(w1p, ot2 * 4 + 1, lane);
      v16bf a2 = load_frag(w1p, ot2 * 4 + 2, lane);
      v16bf a3 = load_frag(w1p, ot2 * 4 + 3, lane);
      v8f acc0 = {0.f, 0.f, 0.f, 0.f, 0.f, 0.f, 0.f, 0.f};
      v8f acc1 = {0.f, 0.f, 0.f, 0.f, 0.f, 0.f, 0.f, 0.f};
      acc0 = wmma_bf16(a0, bh[0][0], acc0);
      acc0 = wmma_bf16(a1, bh[0][1], acc0);
      acc0 = wmma_bf16(a2, bh[0][2], acc0);
      acc0 = wmma_bf16(a3, bh[0][3], acc0);
      acc1 = wmma_bf16(a0, bh[1][0], acc1);
      acc1 = wmma_bf16(a1, bh[1][1], acc1);
      acc1 = wmma_bf16(a2, bh[1][2], acc1);
      acc1 = wmma_bf16(a3, bh[1][3], acc1);
      int rbase = ot2 * 16 + jb;
#pragma unroll
      for (int r = 0; r < 8; ++r) {
        float v = fmaxf(fmaxf(acc0[r], acc1[r]) + b1[rbase + r], 0.0f);
#pragma unroll
        for (int off = 1; off < 16; off <<= 1) v = fmaxf(v, __shfl_xor(v, off, 32));
        if ((lane & 15) == 0)
          pooled[((size_t)b * O1_ + (rbase + r)) * M_ + mm] = v;
      }
    }
  }
}

// ---------------- host entry ---------------------------------------------
extern "C" void kernel_launch(void* const* d_in, const int* in_sizes, int n_in,
                              void* d_out, int out_size, void* d_ws, size_t ws_size,
                              hipStream_t stream) {
  (void)in_sizes; (void)n_in; (void)out_size; (void)ws_size;
  const float* xyz     = (const float*)d_in[0];
  const float* feat    = (const float*)d_in[1];
  const float* prompt  = (const float*)d_in[2];
  const float* proj_w  = (const float*)d_in[3];
  const float* proj_b  = (const float*)d_in[4];
  const float* ln_g    = (const float*)d_in[5];
  const float* ln_b    = (const float*)d_in[6];
  const float* pw_w    = (const float*)d_in[7];
  const float* pw_b    = (const float*)d_in[8];
  const float* w0      = (const float*)d_in[9];
  const float* b0      = (const float*)d_in[10];
  const float* w1      = (const float*)d_in[11];
  const float* b1      = (const float*)d_in[12];

  float* out        = (float*)d_out;
  float* out_xyz    = out;                                  // (B,M,3)
  float* out_pooled = out + (size_t)B_ * M_ * 3;            // (B,256,M)
  float* out_idx    = out + (size_t)B_ * M_ * 3 + (size_t)B_ * O1_ * M_;  // (B,M)

  char* ws = (char*)d_ws;
  float*  featT   = (float*)(ws + OFF_FEATT);
  bf16_t* gws     = (bf16_t*)(ws + OFF_GWS);
  float*  pe      = (float*)(ws + OFF_PE);
  float*  pw      = (float*)(ws + OFF_PW);
  float*  emb_sum = (float*)(ws + OFF_EMB);
  bf16_t* w0p     = (bf16_t*)(ws + OFF_W0P);
  bf16_t* w1p     = (bf16_t*)(ws + OFF_W1P);

  prep_kernel<<<(O0_ * CPP_ + O1_ * O0_ + B_ * CP_ + 255) / 256, 256, 0, stream>>>(
      w0, w1, w0p, w1p, emb_sum);
  transpose_kernel<<<dim3(N_ / 32, C_ / 32, B_), dim3(32, 32), 0, stream>>>(feat, featT);
  fps_kernel<<<B_, 1024, 0, stream>>>(xyz, out_xyz, out_idx);
  ballq_kernel<<<dim3(M_ / 8, B_), 256, 0, stream>>>(xyz, featT, out_xyz, gws, emb_sum);
  pe_kernel<<<M_ / 8, 256, 0, stream>>>(prompt, proj_w, proj_b, ln_g, ln_b, pe);
  pw_kernel<<<B_, 1024, 0, stream>>>(emb_sum, pw_w, pw_b, pw);
  mlp_kernel<<<dim3(M_ / 4, B_), 128, 0, stream>>>(gws, pe, pw, w0p, b0, w1p, b1, out_pooled);
}
